// TFGWPooling_44126493999109
// MI455X (gfx1250) — matrix operand order repaired and checked by reading, not compiled, not run on hardware
//
#include <hip/hip_runtime.h>
#include <hip/hip_bf16.h>
#include <math.h>

// ---------------- problem constants (match reference) ----------------
#define N_NODES    100000
#define N_EDGES    1600000
#define NF         128
#define KT         8        // N_TEMPLATES
#define MT         8        // N_TEMPLATE_NODES
#define KM         64       // KT*MT
#define N_OUTER    5
#define N_SINKHORN 20
#define EPS_REL    0.05f
#define INVN       (1.0f / (float)N_NODES)
#define LOGN       11.512925465f          // log(100000)
#define NEG_BIG    (-3.0e38f)
#define GSPLIT     16                     // blocks per sinkhorn column reduction

typedef __attribute__((ext_vector_type(16))) __bf16 v16bf;
typedef __attribute__((ext_vector_type(8)))  float  v8f;

// smalls[] layout: [0]=alpha, [1..64]=q, [65..128]=logq, [129..192]=h(k,i),
//                  [193..256]=||tf||^2(k,m)

// ---------------- tiny scalar prep (1 block, 64 threads) ----------------
__global__ void prep_small_kernel(const float* __restrict__ templates,
                                  const float* __restrict__ tf,
                                  const float* __restrict__ q0,
                                  const float* __restrict__ alpha0,
                                  float* __restrict__ smalls) {
  const int c = threadIdx.x;            // 0..63 == k*8+i
  if (c >= KM) return;
  const int k = c >> 3, i = c & 7;
  float mx = NEG_BIG;
  for (int j = 0; j < MT; ++j) mx = fmaxf(mx, q0[k * MT + j]);
  float s = 0.f;
  for (int j = 0; j < MT; ++j) s += __expf(q0[k * MT + j] - mx);
  const float logs = __logf(s);
  smalls[1 + c]  = __expf(q0[c] - mx) / s;     // q
  smalls[65 + c] = q0[c] - mx - logs;          // logq
  float h = 0.f;                               // h[k,i] = sum_j T[k,i,j]^2 q[k,j]
  for (int j = 0; j < MT; ++j) {
    float qj = __expf(q0[k * MT + j] - mx) / s;
    float t  = templates[k * 64 + i * 8 + j];
    h += t * t * qj;
  }
  smalls[129 + c] = h;
  float tn = 0.f;                              // ||template_features[k,m]||^2
  for (int f = 0; f < NF; ++f) { float v = tf[c * NF + f]; tn += v * v; }
  smalls[193 + c] = tn;
  if (c == 0) smalls[0] = 1.f / (1.f + __expf(-alpha0[0]));   // sigmoid
}

__global__ void zero_kernel(float* __restrict__ p, int n) {
  int i = blockIdx.x * blockDim.x + threadIdx.x;
  if (i < n) p[i] = 0.f;
}

// C1p[src] += 1/N per edge
__global__ void degree_kernel(const int* __restrict__ esrc, float* __restrict__ C1p) {
  int e = blockIdx.x * blockDim.x + threadIdx.x;
  if (e < N_EDGES) atomicAdd(&C1p[esrc[e]], INVN);
}

__global__ void xnorm_kernel(const float* __restrict__ x, float* __restrict__ xn) {
  int n = blockIdx.x * blockDim.x + threadIdx.x;
  if (n >= N_NODES) return;
  const float4* xp = (const float4*)(x + (size_t)n * NF);
  float s = 0.f;
#pragma unroll
  for (int i = 0; i < NF / 4; ++i) {
    float4 v = xp[i];
    s += v.x * v.x + v.y * v.y + v.z * v.z + v.w * v.w;
  }
  xn[n] = s;
}

// ---------------- WMMA GEMM: G[n, k*8+m] = x[n,:] . tf[k,m,:] ----------------
// bf16 in, f32 accumulate; v_wmma_f32_16x16x32_bf16, K-loop 4x32 over F=128.
// Block = 256 thr (8 waves) -> 64 rows x 64 cols. Wave = (coltile 0..3, rowgroup
// 0..1); each wave runs TWO 16-row strips against ONE shared B fragment ->
// 8 v_wmma per wave, B fragment loaded once per k-step as a single v16bf
// (two ds_load_b128) because B is pre-swizzled into WMMA register order.
__global__ void __launch_bounds__(256) gemm_wmma_kernel(
    const float* __restrict__ x, const float* __restrict__ tf,
    float* __restrict__ G) {
  // Bsh fragment order: [(ct*4 + kk)*32 + lane][e], e=0..15
  //   element e of (kk,lane) = bf16( tf[(ct*16 + lane%16)*128 + kk*32 + 16*(lane>=16) + e] )
  __shared__ __bf16 Bsh[4 * 4 * 32 * 16];   // 16 KB
  const int tid = threadIdx.x;
  for (int idx = tid; idx < 4 * 4 * 32 * 16; idx += 256) {
    int e    = idx & 15;
    int lane = (idx >> 4) & 31;
    int kk   = (idx >> 9) & 3;
    int ct   = idx >> 11;
    int K    = kk * 32 + ((lane & 16) ? 16 : 0) + e;
    int col  = ct * 16 + (lane & 15);
    Bsh[idx] = (__bf16)tf[col * NF + K];
  }
  __syncthreads();
  const int wave = tid >> 5, lane = tid & 31;
  const int ct = wave & 3;              // column tile (16 cols)
  const int rg = wave >> 2;             // row group (32 rows)
  const int col = ct * 16 + (lane & 15);
  const int hiA = (lane >> 4) << 3;     // A K half (hiA<<1 = +16) / D row half (+8)
  const int row0 = blockIdx.x * 64 + rg * 32 + (lane & 15);
  const int row1 = row0 + 16;
  const float* xr0 = x + (size_t)min(row0, N_NODES - 1) * NF;  // clamp tail block
  const float* xr1 = x + (size_t)min(row1, N_NODES - 1) * NF;
  const v16bf* bfrag = ((const v16bf*)Bsh) + (ct * 4) * 32 + lane;
  v8f acc0 = {}, acc1 = {};
#pragma unroll
  for (int kk = 0; kk < 4; ++kk) {
    const int kofs = kk * 32;
    const v16bf b = bfrag[kk * 32];     // one 32B LDS fragment, shared by 2 strips
    v16bf a0, a1;
#pragma unroll
    for (int v = 0; v < 8; ++v) {
      // A: contiguous K order to match B — VGPR v holds K = 2*v + 16*(lane>=16)
      int ka = kofs + (hiA << 1) + (v << 1);
      float2 p0 = *(const float2*)(xr0 + ka);
      float2 p1 = *(const float2*)(xr1 + ka);
      a0[2 * v] = (__bf16)p0.x; a0[2 * v + 1] = (__bf16)p0.y;
      a1[2 * v] = (__bf16)p1.x; a1[2 * v + 1] = (__bf16)p1.y;
    }
    acc0 = __builtin_amdgcn_wmma_f32_16x16x32_bf16(
        false, a0, false, b, (short)0, acc0, false, false);
    acc1 = __builtin_amdgcn_wmma_f32_16x16x32_bf16(
        false, a1, false, b, (short)0, acc1, false, false);
  }
  // D: f32 16x16 — VGPR r holds M = r + 8*(lane>=16), N = lane%16
  const int rb0 = blockIdx.x * 64 + rg * 32 + hiA;
  const int rb1 = rb0 + 16;
#pragma unroll
  for (int r = 0; r < 8; ++r) {
    if (rb0 + r < N_NODES) G[(size_t)(rb0 + r) * KM + col] = acc0[r];
    if (rb1 + r < N_NODES) G[(size_t)(rb1 + r) * KM + col] = acc1[r];
  }
}

// T0[k,n,i] = (1/N) * q[k,i]     (exact grid: N*64 threads)
__global__ void tinit_kernel(float* __restrict__ T, const float* __restrict__ smalls) {
  int idx = blockIdx.x * blockDim.x + threadIdx.x;
  T[idx] = INVN * smalls[1 + (idx & 63)];
}

// CT[src,:] += T[dst,:]  — one wave per edge, 2 cols/lane, float2 gathers
__global__ void spmm_kernel(const int* __restrict__ ei, const float* __restrict__ T,
                            float* __restrict__ CT) {
  int gid = blockIdx.x * blockDim.x + threadIdx.x;
  int e = gid >> 5, lane = gid & 31;
  if (e >= N_EDGES) return;
  if (lane == 0 && e + 4096 < N_EDGES)
    __builtin_prefetch(ei + e + 4096, 0, 1);           // global_prefetch_b8
  int s = ei[e];
  int d = ei[N_EDGES + e];
  float2 v = *(const float2*)(T + (size_t)d * KM + lane * 2);
  atomicAdd(&CT[(size_t)s * KM + lane * 2],     v.x);
  atomicAdd(&CT[(size_t)s * KM + lane * 2 + 1], v.y);
}

// cost = alpha*(C1p[n]+h - 2*CT.tmpl^T) + (1-alpha)*(||x||^2+||tf||^2-2G);
// also accumulates sum|cost| per template into epsacc.
__global__ void __launch_bounds__(256) cost_kernel(
    const float* __restrict__ CT, const float* __restrict__ G,
    const float* __restrict__ xn, const float* __restrict__ C1p,
    const float* __restrict__ templates, const float* __restrict__ smalls,
    float* __restrict__ cost, float* __restrict__ epsacc) {
  __shared__ float tsh[KT * MT * MT];
  __shared__ float hsh[KM], tnsh[KM];
  __shared__ float red[KT * 256];
  const int tid = threadIdx.x;
  for (int i = tid; i < KT * MT * MT; i += 256) tsh[i] = templates[i];
  if (tid < KM) { hsh[tid] = smalls[129 + tid]; tnsh[tid] = smalls[193 + tid]; }
  __syncthreads();
  float acc[KT];
#pragma unroll
  for (int k = 0; k < KT; ++k) acc[k] = 0.f;
  const int n = blockIdx.x * 256 + tid;
  if (n < N_NODES) {
    const float a = smalls[0], bm = 1.f - a;
    const float c1 = C1p[n], x2 = xn[n];
#pragma unroll
    for (int k = 0; k < KT; ++k) {
      const float4* cp = (const float4*)(CT + (size_t)n * KM + k * MT);
      const float4* gp = (const float4*)(G  + (size_t)n * KM + k * MT);
      float4 ca = cp[0], cb = cp[1], ga = gp[0], gb = gp[1];
      float ct8[8] = {ca.x, ca.y, ca.z, ca.w, cb.x, cb.y, cb.z, cb.w};
      float g8[8]  = {ga.x, ga.y, ga.z, ga.w, gb.x, gb.y, gb.z, gb.w};
#pragma unroll
      for (int p = 0; p < MT; ++p) {
        float tens = 0.f;
#pragma unroll
        for (int m = 0; m < MT; ++m) tens += ct8[m] * tsh[k * 64 + p * 8 + m];
        float cv = a  * (c1 + hsh[k * 8 + p] - 2.f * tens)
                 + bm * (x2 + tnsh[k * 8 + p] - 2.f * g8[p]);
        cost[(size_t)n * KM + k * MT + p] = cv;
        acc[k] += fabsf(cv);
      }
    }
  }
#pragma unroll
  for (int k = 0; k < KT; ++k) red[k * 256 + tid] = acc[k];
  __syncthreads();
  for (int s = 128; s > 0; s >>= 1) {
    if (tid < s) {
#pragma unroll
      for (int k = 0; k < KT; ++k) red[k * 256 + tid] += red[k * 256 + tid + s];
    }
    __syncthreads();
  }
  if (tid < KT) atomicAdd(&epsacc[tid], red[tid * 256]);
}

__global__ void eps_finalize_kernel(const float* __restrict__ epsacc,
                                    float* __restrict__ eps, float* __restrict__ g) {
  int t = threadIdx.x;
  if (t < KT) eps[t] = EPS_REL * epsacc[t] * (1.f / ((float)N_NODES * MT));
  if (t < KM) g[t] = 0.f;
}

// g-update, phase 1: online logsumexp over a node chunk, per cost column c.
__global__ void __launch_bounds__(256) sinkhorn_g_partial_kernel(
    const float* __restrict__ cost, const float* __restrict__ f,
    const float* __restrict__ eps, float* __restrict__ pmx, float* __restrict__ psm) {
  const int c = blockIdx.x;            // 0..63
  const int k = c >> 3;
  const int seg = blockIdx.y;          // 0..GSPLIT-1
  const int tid = threadIdx.x;
  const float inve = 1.f / eps[k];
  const int chunk = (N_NODES + GSPLIT - 1) / GSPLIT;
  const int n0 = seg * chunk;
  const int n1 = min(N_NODES, n0 + chunk);
  const float* fk = f + (size_t)k * N_NODES;
  float mx = NEG_BIG, sm = 0.f;
  for (int n = n0 + tid; n < n1; n += 256) {
    float v = (fk[n] - cost[(size_t)n * KM + c]) * inve;
    if (v > mx) { sm = sm * __expf(mx - v) + 1.f; mx = v; }
    else        { sm += __expf(v - mx); }
  }
  __shared__ float smx[256], ssm[256];
  smx[tid] = mx; ssm[tid] = sm;
  __syncthreads();
  for (int s = 128; s > 0; s >>= 1) {
    if (tid < s) {
      float m1 = smx[tid],     s1 = ssm[tid];
      float m2 = smx[tid + s], s2 = ssm[tid + s];
      if (m2 > m1) { smx[tid] = m2; ssm[tid] = s1 * __expf(m1 - m2) + s2; }
      else         { ssm[tid] = s1 + s2 * __expf(m2 - m1); }
    }
    __syncthreads();
  }
  if (tid == 0) { pmx[c * GSPLIT + seg] = smx[0]; psm[c * GSPLIT + seg] = ssm[0]; }
}

// g-update, phase 2: combine partials, g = e*logq - e*lse
__global__ void sinkhorn_g_finalize_kernel(
    const float* __restrict__ pmx, const float* __restrict__ psm,
    const float* __restrict__ eps, const float* __restrict__ smalls,
    float* __restrict__ g) {
  int c = threadIdx.x;
  if (c >= KM) return;
  float mx = NEG_BIG, sm = 0.f;
  for (int s = 0; s < GSPLIT; ++s) {
    float m2 = pmx[c * GSPLIT + s], s2 = psm[c * GSPLIT + s];
    if (m2 > mx) { sm = sm * __expf(mx - m2) + s2; mx = m2; }
    else         { sm += s2 * __expf(m2 - mx); }
  }
  float e = eps[c >> 3];
  g[c] = e * smalls[65 + c] - e * (mx + __logf(sm));
}

// f-update: per (k,n), reduce over m=8. f = -e*logN - e*lse
__global__ void sinkhorn_f_kernel(const float* __restrict__ cost,
                                  const float* __restrict__ g,
                                  const float* __restrict__ eps,
                                  float* __restrict__ f) {
  int gid = blockIdx.x * blockDim.x + threadIdx.x;
  if (gid >= KT * N_NODES) return;
  int k = gid / N_NODES;
  int n = gid - k * N_NODES;
  float e = eps[k], inve = 1.f / e;
  const float4* cp = (const float4*)(cost + (size_t)n * KM + k * MT);
  float4 ca = cp[0], cb = cp[1];
  float cv[8] = {ca.x, ca.y, ca.z, ca.w, cb.x, cb.y, cb.z, cb.w};
  float v[8], mx = NEG_BIG;
#pragma unroll
  for (int i = 0; i < MT; ++i) { v[i] = (g[k * MT + i] - cv[i]) * inve; mx = fmaxf(mx, v[i]); }
  float s = 0.f;
#pragma unroll
  for (int i = 0; i < MT; ++i) s += __expf(v[i] - mx);
  f[gid] = -e * LOGN - e * (mx + __logf(s));
}

// T = exp((f + g - cost)/eps)        (exact grid: N*64 threads)
__global__ void t_update_kernel(const float* __restrict__ cost,
                                const float* __restrict__ f,
                                const float* __restrict__ g,
                                const float* __restrict__ eps,
                                float* __restrict__ T) {
  int idx = blockIdx.x * blockDim.x + threadIdx.x;
  int c = idx & 63, n = idx >> 6, k = c >> 3;
  T[idx] = __expf((f[(size_t)k * N_NODES + n] + g[c] - cost[idx]) / eps[k]);
}

// fgw[k] = sum_{n,p} ((1-a)*Mfeat + a*tens) * T
__global__ void __launch_bounds__(256) final_kernel(
    const float* __restrict__ CT, const float* __restrict__ G,
    const float* __restrict__ xn, const float* __restrict__ C1p,
    const float* __restrict__ templates, const float* __restrict__ smalls,
    const float* __restrict__ T, float* __restrict__ out) {
  __shared__ float tsh[KT * MT * MT];
  __shared__ float hsh[KM], tnsh[KM];
  __shared__ float red[KT * 256];
  const int tid = threadIdx.x;
  for (int i = tid; i < KT * MT * MT; i += 256) tsh[i] = templates[i];
  if (tid < KM) { hsh[tid] = smalls[129 + tid]; tnsh[tid] = smalls[193 + tid]; }
  __syncthreads();
  float acc[KT];
#pragma unroll
  for (int k = 0; k < KT; ++k) acc[k] = 0.f;
  const int n = blockIdx.x * 256 + tid;
  if (n < N_NODES) {
    const float a = smalls[0], bm = 1.f - a;
    const float c1 = C1p[n], x2 = xn[n];
#pragma unroll
    for (int k = 0; k < KT; ++k) {
      const float4* cp = (const float4*)(CT + (size_t)n * KM + k * MT);
      const float4* gp = (const float4*)(G  + (size_t)n * KM + k * MT);
      const float4* tp = (const float4*)(T  + (size_t)n * KM + k * MT);
      float4 ca = cp[0], cb = cp[1], ga = gp[0], gb = gp[1], ta = tp[0], tb = tp[1];
      float ct8[8] = {ca.x, ca.y, ca.z, ca.w, cb.x, cb.y, cb.z, cb.w};
      float g8[8]  = {ga.x, ga.y, ga.z, ga.w, gb.x, gb.y, gb.z, gb.w};
      float t8[8]  = {ta.x, ta.y, ta.z, ta.w, tb.x, tb.y, tb.z, tb.w};
#pragma unroll
      for (int p = 0; p < MT; ++p) {
        float tens = 0.f;
#pragma unroll
        for (int m = 0; m < MT; ++m) tens += ct8[m] * tsh[k * 64 + p * 8 + m];
        float gw = c1 + hsh[k * 8 + p] - 2.f * tens;
        float mf = x2 + tnsh[k * 8 + p] - 2.f * g8[p];
        acc[k] += (bm * mf + a * gw) * t8[p];
      }
    }
  }
#pragma unroll
  for (int k = 0; k < KT; ++k) red[k * 256 + tid] = acc[k];
  __syncthreads();
  for (int s = 128; s > 0; s >>= 1) {
    if (tid < s) {
#pragma unroll
      for (int k = 0; k < KT; ++k) red[k * 256 + tid] += red[k * 256 + tid + s];
    }
    __syncthreads();
  }
  if (tid < KT) atomicAdd(&out[tid], red[tid * 256]);
}

// ---------------- host orchestration (graph-capture safe) ----------------
extern "C" void kernel_launch(void* const* d_in, const int* in_sizes, int n_in,
                              void* d_out, int out_size, void* d_ws, size_t ws_size,
                              hipStream_t stream) {
  (void)in_sizes; (void)n_in; (void)out_size; (void)ws_size;
  const float* x   = (const float*)d_in[0];
  const int*   ei  = (const int*)d_in[1];      // (2, E): [0..E)=src, [E..2E)=dst
  const float* tpl = (const float*)d_in[2];
  const float* tf  = (const float*)d_in[3];
  const float* q0  = (const float*)d_in[4];
  const float* a0  = (const float*)d_in[5];
  float* out = (float*)d_out;
  float* ws  = (float*)d_ws;

  // workspace carve-out (~107 MB of f32; assumes ws_size is sufficient)
  const size_t NE = (size_t)N_NODES * KM;
  float* G      = ws;
  float* Tm     = G + NE;
  float* CT     = Tm + NE;
  float* cost   = CT + NE;
  float* xnorm  = cost + NE;
  float* C1p    = xnorm + N_NODES;
  float* fmat   = C1p + N_NODES;                 // K x N
  float* smalls = fmat + (size_t)KT * N_NODES;   // 257 used
  float* g      = smalls + 260;
  float* eps    = g + KM;
  float* epsacc = eps + KT;
  float* pmx    = epsacc + KT;                   // 64*GSPLIT
  float* psm    = pmx + KM * GSPLIT;

  const dim3 b256(256);
  const int nb_nodes = (N_NODES + 255) / 256;        // 391
  const int nb_edges = (N_EDGES + 255) / 256;        // 6250
  const int nb_plane = (int)(NE / 256);              // 25000 (exact)
  const int nb_spmm  = (N_EDGES * 32) / 256;         // 200000 (exact)
  const int nb_kn    = (KT * N_NODES + 255) / 256;   // 3125 (exact)
  const int nb_gemm  = (N_NODES + 63) / 64;          // 1563 (tail clamped)

  prep_small_kernel<<<1, 64, 0, stream>>>(tpl, tf, q0, a0, smalls);
  zero_kernel<<<nb_nodes, b256, 0, stream>>>(C1p, N_NODES);
  degree_kernel<<<nb_edges, b256, 0, stream>>>(ei, C1p);
  xnorm_kernel<<<nb_nodes, b256, 0, stream>>>(x, xnorm);
  gemm_wmma_kernel<<<nb_gemm, b256, 0, stream>>>(x, tf, G);
  tinit_kernel<<<nb_plane, b256, 0, stream>>>(Tm, smalls);

  for (int it = 0; it < N_OUTER; ++it) {
    zero_kernel<<<nb_plane, b256, 0, stream>>>(CT, (int)NE);
    spmm_kernel<<<nb_spmm, b256, 0, stream>>>(ei, Tm, CT);
    zero_kernel<<<1, b256, 0, stream>>>(epsacc, KT);
    cost_kernel<<<nb_nodes, b256, 0, stream>>>(CT, G, xnorm, C1p, tpl, smalls,
                                               cost, epsacc);
    eps_finalize_kernel<<<1, 64, 0, stream>>>(epsacc, eps, g);
    zero_kernel<<<nb_kn, b256, 0, stream>>>(fmat, KT * N_NODES);
    for (int s = 0; s < N_SINKHORN; ++s) {
      sinkhorn_g_partial_kernel<<<dim3(KM, GSPLIT), b256, 0, stream>>>(
          cost, fmat, eps, pmx, psm);
      sinkhorn_g_finalize_kernel<<<1, 64, 0, stream>>>(pmx, psm, eps, smalls, g);
      sinkhorn_f_kernel<<<nb_kn, b256, 0, stream>>>(cost, g, eps, fmat);
    }
    t_update_kernel<<<nb_plane, b256, 0, stream>>>(cost, fmat, g, eps, Tm);
  }

  zero_kernel<<<nb_plane, b256, 0, stream>>>(CT, (int)NE);
  spmm_kernel<<<nb_spmm, b256, 0, stream>>>(ei, Tm, CT);
  zero_kernel<<<1, 64, 0, stream>>>(out, KT);
  final_kernel<<<nb_nodes, b256, 0, stream>>>(CT, G, xnorm, C1p, tpl, smalls,
                                              Tm, out);
}
